// QuantumAI5Qubits_61375082660339
// MI455X (gfx1250) — matrix operand order repaired and checked by reading, not compiled, not run on hardware
//
#include <hip/hip_runtime.h>
#include <hip/hip_bf16.h>
#include <math.h>

typedef __attribute__((ext_vector_type(16))) __bf16 v16bf;
typedef __attribute__((ext_vector_type(8)))  float  v8f;

union BF16x16 { v16bf v; unsigned u[8]; };

__device__ __forceinline__ unsigned short f2bf(float x) {
  union { float f; unsigned u; } cv; cv.f = x;
  unsigned u = cv.u;
  unsigned r = (u + 0x7FFFu + ((u >> 16) & 1u)) >> 16;   // round-to-nearest-even
  return (unsigned short)r;
}
__device__ __forceinline__ unsigned pack2(float a, float b) {
  return (unsigned)f2bf(a) | ((unsigned)f2bf(b) << 16);
}

#define S_IN 252
#define KPAD 256

__global__ __launch_bounds__(128, 2)
void qai5_fused(const float* __restrict__ state,
                const float* __restrict__ W1, const float* __restrict__ b1,
                const float* __restrict__ W2, const float* __restrict__ b2,
                const float* __restrict__ W3, const float* __restrict__ b3,
                const float* __restrict__ qp,
                const float* __restrict__ D1, const float* __restrict__ d1,
                const float* __restrict__ D2, const float* __restrict__ d2,
                const float* __restrict__ D3, const float* __restrict__ d3,
                float* __restrict__ out)
{
  // ---- LDS ----
  __shared__ unsigned short w1t[128 * KPAD];     // W1^T bf16  [n][k]   64 KB
  __shared__ unsigned short w2t[64 * 128];       // W2^T bf16  [n][k]   16 KB
  __shared__ float w3s[64 * 5];
  __shared__ float b1s[128], b2s[64], b3s[5];
  __shared__ float d1w[5 * 32], d1b[32], d2w[32 * 16], d2b[16], d3w[16 * 4], d3b[4];
  __shared__ unsigned short h1s[4][16 * 128];    // per-wave h1 bf16    16 KB
  __shared__ float          h2s[4][16 * 64];     // per-wave h2 f32     16 KB
  __shared__ float          encS[4][16 * 5];

  const int tid  = threadIdx.x;
  const int lane = tid & 31;
  const int wave = tid >> 5;
  const int half = lane >> 4;   // WMMA lane half
  const int l16  = lane & 15;

  // ---- cooperative weight conversion into LDS ----
  for (int idx = tid; idx < 128 * KPAD; idx += 128) {
    int n = idx & 127, k = idx >> 7;          // consecutive tid -> consecutive n (coalesced read)
    w1t[n * KPAD + k] = (k < S_IN) ? f2bf(W1[k * 128 + n]) : (unsigned short)0;
  }
  for (int idx = tid; idx < 64 * 128; idx += 128) {
    int n = idx & 63, k = idx >> 6;
    w2t[n * 128 + k] = f2bf(W2[k * 64 + n]);
  }
  for (int idx = tid; idx < 320; idx += 128) w3s[idx] = W3[idx];
  if (tid < 128) b1s[tid] = b1[tid];
  if (tid < 64)  b2s[tid] = b2[tid];
  if (tid < 5)   b3s[tid] = b3[tid];
  for (int idx = tid; idx < 160; idx += 128) d1w[idx] = D1[idx];
  if (tid < 32) d1b[tid] = d1[tid];
  for (int idx = tid; idx < 512; idx += 128) d2w[idx] = D2[idx];
  if (tid < 16) d2b[tid] = d2[tid];
  if (tid < 64) d3w[tid] = D3[tid];
  if (tid < 4)  d3b[tid] = d3[tid];
  __syncthreads();

  const long rowBase = (long)blockIdx.x * 64 + wave * 16;
  const float* aRow = state + (rowBase + l16) * S_IN;

  // ======== GEMM1: state(16x252) @ W1 -> h1(16x128), bf16 WMMA, f32 acc ========
  v8f acc[8] = {};

  // A fragment loader (16-bit A layout: VGPR v<4: K = base + half*8 + 2v ; v>=4: +16)
  auto loadA = [&](int k0, bool tail) -> BF16x16 {
    BF16x16 A;
    #pragma unroll
    for (int v = 0; v < 8; ++v) {
      int k = k0 * 32 + (v >> 2) * 16 + half * 8 + (v & 3) * 2;
      float x0 = 0.f, x1 = 0.f;
      if (!tail || k < S_IN) {                 // folds to unconditional when tail==false
        float2 t = *reinterpret_cast<const float2*>(aRow + k);
        x0 = t.x; x1 = t.y;
      }
      A.u[v] = pack2(x0, x1);
    }
    return A;
  };
  // 8 WMMAs against one K-chunk of W1^T (B layout: VGPR v<4: K = base + half*8 + 2v ; v>=4: +16)
  auto wmmaBlock1 = [&](const BF16x16& A, int k0) {
    #pragma unroll
    for (int nt = 0; nt < 8; ++nt) {
      BF16x16 Bf;
      #pragma unroll
      for (int v = 0; v < 8; ++v) {
        int k   = k0 * 32 + (v >> 2) * 16 + half * 8 + (v & 3) * 2;
        int col = nt * 16 + l16;
        Bf.u[v] = *reinterpret_cast<const unsigned*>(&w1t[col * KPAD + k]);
      }
      acc[nt] = __builtin_amdgcn_wmma_f32_16x16x32_bf16(false, A.v, false, Bf.v,
                                                        (short)0, acc[nt], false, false);
    }
  };

  // software pipeline: process chunk k0 while chunk k0+1 streams from HBM
  BF16x16 Acur = loadA(0, false);
  #pragma unroll 1
  for (int k0 = 0; k0 < 6; ++k0) {
    BF16x16 Anxt = loadA(k0 + 1, false);     // chunks 1..6 are fully in-bounds (k <= 222)
    wmmaBlock1(Acur, k0);
    Acur = Anxt;
  }
  BF16x16 Atail = loadA(7, true);            // only chunk 7 carries the k<252 mask
  wmmaBlock1(Acur, 6);
  wmmaBlock1(Atail, 7);

  // bias + relu -> h1 staged bf16 (C layout: VGPR i -> M = half*8+i, N = lane&15)
  #pragma unroll
  for (int nt = 0; nt < 8; ++nt) {
    #pragma unroll
    for (int i = 0; i < 8; ++i) {
      int m = half * 8 + i, col = nt * 16 + l16;
      float v = acc[nt][i] + b1s[col];
      h1s[wave][m * 128 + col] = f2bf(v > 0.f ? v : 0.f);
    }
  }

  // ======== GEMM2: h1(16x128) @ W2 -> h2(16x64) ========
  v8f acc2[4] = {};
  for (int k0 = 0; k0 < 4; ++k0) {
    BF16x16 A;
    #pragma unroll
    for (int v = 0; v < 8; ++v) {
      int k = k0 * 32 + (v >> 2) * 16 + half * 8 + (v & 3) * 2;
      A.u[v] = *reinterpret_cast<const unsigned*>(&h1s[wave][l16 * 128 + k]);
    }
    #pragma unroll
    for (int nt = 0; nt < 4; ++nt) {
      BF16x16 Bf;
      #pragma unroll
      for (int v = 0; v < 8; ++v) {
        int k   = k0 * 32 + (v >> 2) * 16 + half * 8 + (v & 3) * 2;
        int col = nt * 16 + l16;
        Bf.u[v] = *reinterpret_cast<const unsigned*>(&w2t[col * 128 + k]);
      }
      acc2[nt] = __builtin_amdgcn_wmma_f32_16x16x32_bf16(false, A.v, false, Bf.v,
                                                         (short)0, acc2[nt], false, false);
    }
  }
  #pragma unroll
  for (int nt = 0; nt < 4; ++nt) {
    #pragma unroll
    for (int i = 0; i < 8; ++i) {
      int m = half * 8 + i, col = nt * 16 + l16;
      float v = acc2[nt][i] + b2s[col];
      h2s[wave][m * 64 + col] = v > 0.f ? v : 0.f;
    }
  }

  // ======== GEMM3 + tanh -> enc (lanes 0-15, one row each) ========
  if (lane < 16) {
    #pragma unroll
    for (int j = 0; j < 5; ++j) {
      float s = b3s[j];
      for (int k = 0; k < 64; ++k) s += h2s[wave][lane * 64 + k] * w3s[k * 5 + j];
      encS[wave][lane * 5 + j] = tanhf(s);
    }
  }

  // ======== quantum circuit: amplitude-per-lane (32 amps == wave32) ========
  // qubit wire w lives at statevector index bit (4-w)
  float cry[15], sry[15], crz[15], srz[15];
  #pragma unroll
  for (int i = 0; i < 15; ++i) {
    float hy = qp[2 * i] * 0.5f, hz = qp[2 * i + 1] * 0.5f;
    cry[i] = cosf(hy); sry[i] = sinf(hy);
    crz[i] = cosf(hz); srz[i] = sinf(hz);
  }

  float qv[5] = {0.f, 0.f, 0.f, 0.f, 0.f};
  for (int r = 0; r < 16; ++r) {
    float cw[5], sw[5];
    #pragma unroll
    for (int w = 0; w < 5; ++w) {
      float hh = encS[wave][r * 5 + w] * 1.5707963267948966f;  // enc*pi/2
      cw[w] = cosf(hh); sw[w] = sinf(hh);
    }
    float re = (lane == 0) ? 1.f : 0.f, im = 0.f;
    // encoding RYs
    #pragma unroll
    for (int w = 0; w < 5; ++w) {
      int bp = 4 - w;
      float pre = __shfl_xor(re, 1 << bp, 32);
      float pim = __shfl_xor(im, 1 << bp, 32);
      float sg  = ((lane >> bp) & 1) ? sw[w] : -sw[w];
      re = cw[w] * re + sg * pre;
      im = cw[w] * im + sg * pim;
    }
    // variational layers
    #pragma unroll
    for (int l = 0; l < 3; ++l) {
      #pragma unroll
      for (int w = 0; w < 5; ++w) {
        int i = l * 5 + w, bp = 4 - w;
        float pre = __shfl_xor(re, 1 << bp, 32);
        float pim = __shfl_xor(im, 1 << bp, 32);
        float sg  = ((lane >> bp) & 1) ? sry[i] : -sry[i];
        re = cry[i] * re + sg * pre;
        im = cry[i] * im + sg * pim;
        float sz  = ((lane >> bp) & 1) ? srz[i] : -srz[i];      // RZ: e^{±i h}
        float nre = re * crz[i] - sz * im;
        float nim = im * crz[i] + sz * re;
        re = nre; im = nim;
      }
      #pragma unroll
      for (int c = 0; c < 4; ++c) {                              // CNOT ladder
        int tb = 3 - c, cb = 4 - c;
        float pre = __shfl_xor(re, 1 << tb, 32);
        float pim = __shfl_xor(im, 1 << tb, 32);
        bool ctrl = (lane >> cb) & 1;
        re = ctrl ? pre : re;
        im = ctrl ? pim : im;
      }
    }
    // <Z_w> expvals via butterfly reduction; lane r keeps row r's values
    float p = re * re + im * im;
    #pragma unroll
    for (int w = 0; w < 5; ++w) {
      float v = ((lane >> (4 - w)) & 1) ? -p : p;
      #pragma unroll
      for (int off = 16; off >= 1; off >>= 1) v += __shfl_xor(v, off, 32);
      if (lane == r) qv[w] = v;
    }
  }

  // ======== tail MLP (lanes 0-15, one row each) ========
  if (lane < 16) {
    float h[32];
    #pragma unroll
    for (int j = 0; j < 32; ++j) {
      float s = d1b[j];
      #pragma unroll
      for (int w = 0; w < 5; ++w) s += qv[w] * d1w[w * 32 + j];
      h[j] = s > 0.f ? s : 0.f;
    }
    float g[16];
    #pragma unroll
    for (int j = 0; j < 16; ++j) {
      float s = d2b[j];
      #pragma unroll
      for (int k = 0; k < 32; ++k) s += h[k] * d2w[k * 16 + j];
      g[j] = s > 0.f ? s : 0.f;
    }
    float* op = out + (rowBase + lane) * 4;
    #pragma unroll
    for (int j = 0; j < 4; ++j) {
      float s = d3b[j];
      #pragma unroll
      for (int k = 0; k < 16; ++k) s += g[k] * d3w[k * 4 + j];
      op[j] = s;
    }
  }
}

extern "C" void kernel_launch(void* const* d_in, const int* in_sizes, int n_in,
                              void* d_out, int out_size, void* d_ws, size_t ws_size,
                              hipStream_t stream) {
  (void)n_in; (void)d_ws; (void)ws_size; (void)out_size;
  const float* state = (const float*)d_in[0];
  const float* W1 = (const float*)d_in[1];  const float* b1 = (const float*)d_in[2];
  const float* W2 = (const float*)d_in[3];  const float* b2 = (const float*)d_in[4];
  const float* W3 = (const float*)d_in[5];  const float* b3 = (const float*)d_in[6];
  const float* qp = (const float*)d_in[7];
  const float* D1 = (const float*)d_in[8];  const float* d1 = (const float*)d_in[9];
  const float* D2 = (const float*)d_in[10]; const float* d2 = (const float*)d_in[11];
  const float* D3 = (const float*)d_in[12]; const float* d3 = (const float*)d_in[13];

  int B = in_sizes[0] / S_IN;          // 131072
  int blocks = B / 64;                 // 64 rows per 128-thread block (4 waves x 16 rows)
  qai5_fused<<<blocks, 128, 0, stream>>>(state, W1, b1, W2, b2, W3, b3, qp,
                                         D1, d1, D2, d2, D3, d3, (float*)d_out);
}
